// DecoderRNN_83966610637069
// MI455X (gfx1250) — matrix operand order repaired and checked by reading, not compiled, not run on hardware
//
#include <hip/hip_runtime.h>
#include <math.h>

// ---------------------------------------------------------------------------
// DecoderRNN greedy decode, CDNA5 (gfx1250, wave32, WMMA).
//
// Shapes: BATCH=32, EMBED=256, HIDDEN=512, VOCAB=32000, MAX_LEN=64.
// Dominant cost: W_out re-read each step. fp32 = 65.5MB*64 = 4.2GB HBM.
// We convert weights to f16 once -> W_out = 32.8MB, fits MI455X 192MB L2,
// so steps 1..63 stream W_out from L2, and matmuls run on v_wmma_f32_16x16x32_f16.
// ---------------------------------------------------------------------------

#define BATCH   32
#define EMBED   256
#define HIDDEN  512
#define VOCAB   32000
#define MAX_LEN 64
#define GO_ID   2
#define EOS_ID  3

typedef __attribute__((ext_vector_type(16))) _Float16 v16h;
typedef __attribute__((ext_vector_type(8)))  float    v8f;

// ---- WMMA fragment helpers (contiguous k order shared by A and B, wave32) ----------

// A fragment: 16x32 f16 tile, source is row-major f32 [*, ld].
// lanes 0-15: row = lane,  halves 0..15 = K 0..15
// lanes16-31: row = lane-16, halves 0..15 = K 16..31
__device__ __forceinline__ v16h load_a_frag_f32(const float* __restrict__ src,
                                                int ld, int rowBase, int kBase,
                                                int lane) {
  const float* p = src + (rowBase + (lane & 15)) * ld + kBase + ((lane < 16) ? 0 : 16);
  v16h a;
#pragma unroll
  for (int i = 0; i < 8; ++i) {
    a[i]     = (_Float16)p[i];
    a[i + 8] = (_Float16)p[i + 8];
  }
  return a;
}

// B fragment: 32x16 f16 tile where B[k][n] = W[colBase+n][kBase+k]
// (i.e. computing X @ W^T with W row-major [N, ld]).
// lanes 0-15: col = lane,  halves 0..15 = K 0..15
// lanes16-31: col = lane-16, halves 0..15 = K 16..31
// -> contiguous 32-byte load per lane from one W row.
__device__ __forceinline__ v16h load_b_frag_f16(const _Float16* __restrict__ W,
                                                int ld, int colBase, int kBase,
                                                int lane) {
  const _Float16* p = W + (colBase + (lane & 15)) * ld + kBase + ((lane < 16) ? 0 : 16);
  v16h b;
#pragma unroll
  for (int i = 0; i < 16; ++i) b[i] = p[i];
  return b;
}

__device__ __forceinline__ v8f wmma_f16(v16h a, v16h b, v8f c) {
  return __builtin_amdgcn_wmma_f32_16x16x32_f16(
      /*neg_a=*/false, a, /*neg_b=*/false, b,
      /*c_mod=*/(short)0, c, /*reuse_a=*/false, /*reuse_b=*/false);
}

__device__ __forceinline__ float sigmoidf_(float x) {
  return 1.0f / (1.0f + expf(-x));
}

// ---------------------------------------------------------------------------
// Prep: fp32 -> fp16 weight conversion (once per launch, grid-stride).
// ---------------------------------------------------------------------------
__global__ void k_f32_to_f16(const float* __restrict__ src,
                             _Float16* __restrict__ dst, int n) {
  for (int i = blockIdx.x * blockDim.x + threadIdx.x; i < n;
       i += gridDim.x * blockDim.x)
    dst[i] = (_Float16)src[i];
}

// Prep: initial state. hbuf0 = h0; ebuf0 = broadcast(emb_table[GO_ID]); flags=0.
__global__ void k_init_state(const float* __restrict__ h0,
                             const float* __restrict__ emb_table,
                             float* __restrict__ hbuf,
                             float* __restrict__ ebuf,
                             int* __restrict__ flags) {
  int i = blockIdx.x * blockDim.x + threadIdx.x;
  if (i < BATCH * HIDDEN) hbuf[i] = h0[i];
  if (i < BATCH * EMBED)  ebuf[i] = emb_table[GO_ID * EMBED + (i & (EMBED - 1))];
  if (i < BATCH)          flags[i] = 0;
}

// ---------------------------------------------------------------------------
// GRU cell. grid = HIDDEN/16 = 32 blocks, 64 threads (2 waves: row tiles).
// Each block owns hidden units [16*blockIdx.x, +16) across all 3 gates.
// acc_r, acc_z accumulate i+h fused; n-gate needs i_n and h_n separate.
// ---------------------------------------------------------------------------
__global__ __launch_bounds__(64) void k_gru_cell(
    const float* __restrict__ x,        // [32,256]  f32 (prev embedding)
    const float* __restrict__ hin,      // [32,512]  f32
    const _Float16* __restrict__ Wih,   // [1536,256] f16
    const _Float16* __restrict__ Whh,   // [1536,512] f16
    const float* __restrict__ bih,      // [1536]
    const float* __restrict__ bhh,      // [1536]
    float* __restrict__ hout)           // [32,512]
{
  const int lane    = threadIdx.x & 31;
  const int wave    = threadIdx.x >> 5;   // 0,1 -> row tile
  const int rowBase = wave * 16;
  const int jBase   = blockIdx.x * 16;

  v8f acc_r  = {}, acc_z = {}, acc_ni = {}, acc_nh = {};

  // x @ W_ih^T contribution (K = 256)
  for (int k = 0; k < EMBED; k += 32) {
    v16h a  = load_a_frag_f32(x, EMBED, rowBase, k, lane);
    v16h br = load_b_frag_f16(Wih, EMBED, 0 * HIDDEN + jBase, k, lane);
    v16h bz = load_b_frag_f16(Wih, EMBED, 1 * HIDDEN + jBase, k, lane);
    v16h bn = load_b_frag_f16(Wih, EMBED, 2 * HIDDEN + jBase, k, lane);
    acc_r  = wmma_f16(a, br, acc_r);
    acc_z  = wmma_f16(a, bz, acc_z);
    acc_ni = wmma_f16(a, bn, acc_ni);
  }
  // h @ W_hh^T contribution (K = 512)
  for (int k = 0; k < HIDDEN; k += 32) {
    v16h a  = load_a_frag_f32(hin, HIDDEN, rowBase, k, lane);
    v16h br = load_b_frag_f16(Whh, HIDDEN, 0 * HIDDEN + jBase, k, lane);
    v16h bz = load_b_frag_f16(Whh, HIDDEN, 1 * HIDDEN + jBase, k, lane);
    v16h bn = load_b_frag_f16(Whh, HIDDEN, 2 * HIDDEN + jBase, k, lane);
    acc_r  = wmma_f16(a, br, acc_r);
    acc_z  = wmma_f16(a, bz, acc_z);
    acc_nh = wmma_f16(a, bn, acc_nh);
  }

  // Epilogue. C/D layout: VGPR r -> M = rowBase + r (+8 for lanes 16-31),
  // N = jBase + (lane & 15).
  const int j    = jBase + (lane & 15);
  const int mAdd = (lane < 16) ? 0 : 8;
  const float b_r = bih[j] + bhh[j];
  const float b_z = bih[HIDDEN + j] + bhh[HIDDEN + j];
  const float b_in = bih[2 * HIDDEN + j];
  const float b_hn = bhh[2 * HIDDEN + j];
#pragma unroll
  for (int r = 0; r < 8; ++r) {
    const int m = rowBase + mAdd + r;
    float rg = sigmoidf_(acc_r[r] + b_r);
    float zg = sigmoidf_(acc_z[r] + b_z);
    float ng = tanhf((acc_ni[r] + b_in) + rg * (acc_nh[r] + b_hn));
    float hp = hin[m * HIDDEN + j];
    hout[m * HIDDEN + j] = (1.0f - zg) * ng + zg * hp;
  }
}

// ---------------------------------------------------------------------------
// Logits: [32,512] @ [512,32000] via f16 WMMA, f32 accumulate.
// grid = VOCAB/64 = 500 blocks, 64 threads (2 waves). Each block computes a
// 32x64 tile; each A fragment is reused across 4 WMMAs.
// ---------------------------------------------------------------------------
__global__ __launch_bounds__(64) void k_logits(
    const float* __restrict__ h,        // [32,512]
    const _Float16* __restrict__ Wout,  // [32000,512] f16
    const float* __restrict__ bout,     // [32000]
    float* __restrict__ logits)         // [32,32000]
{
  const int lane    = threadIdx.x & 31;
  const int wave    = threadIdx.x >> 5;
  const int rowBase = wave * 16;
  const int nBase   = blockIdx.x * 64;

  v8f acc[4] = {{}, {}, {}, {}};
  for (int k = 0; k < HIDDEN; k += 32) {
    // warm L0/L2 ahead (emits global_prefetch_b8)
    if (k + 64 < HIDDEN)
      __builtin_prefetch(Wout + (size_t)(nBase + (lane & 15)) * HIDDEN + k + 64, 0, 1);
    v16h a = load_a_frag_f32(h, HIDDEN, rowBase, k, lane);
#pragma unroll
    for (int t = 0; t < 4; ++t) {
      v16h b = load_b_frag_f16(Wout, HIDDEN, nBase + 16 * t, k, lane);
      acc[t] = wmma_f16(a, b, acc[t]);
    }
  }

  const int mAdd = (lane < 16) ? 0 : 8;
#pragma unroll
  for (int t = 0; t < 4; ++t) {
    const int col = nBase + 16 * t + (lane & 15);
    const float bb = bout[col];
#pragma unroll
    for (int r = 0; r < 8; ++r) {
      const int m = rowBase + mAdd + r;
      logits[(size_t)m * VOCAB + col] = acc[t][r] + bb;
    }
  }
}

// ---------------------------------------------------------------------------
// Softmax + argmax(>=GO_ID) + EOS flags + embedding gather.
// One block (256 threads) per batch row. LDS tree reductions; argmax keeps
// the lowest index on ties (jnp.argmax semantics).
// ---------------------------------------------------------------------------
__global__ __launch_bounds__(256) void k_softmax_argmax(
    const float* __restrict__ logits,     // [32,32000]
    const float* __restrict__ emb_table,  // [32000,256]
    float* __restrict__ w_pro_out,        // [32,32000] slice of d_out
    float* __restrict__ w_o_out,          // [32]       slice of d_out
    float* __restrict__ emb_seq_out,      // [32,256]   slice of d_out
    float* __restrict__ next_emb,         // [32,256]   ws (unmasked carry)
    int* __restrict__ flags,              // [32]       persistent across steps
    int* __restrict__ alive_out)          // [32]       slice of alive[64][32]
{
  const int b   = blockIdx.x;
  const int tid = threadIdx.x;
  const float* row = logits + (size_t)b * VOCAB;

  __shared__ float s_max[256];
  __shared__ float s_amax[256];
  __shared__ int   s_aidx[256];

  // Pass 1: global max (all cols) + argmax over cols >= GO_ID.
  float lmax = -INFINITY, lamax = -INFINITY;
  int laidx = VOCAB;
  for (int i = tid; i < VOCAB; i += 256) {
    float v = row[i];
    lmax = fmaxf(lmax, v);
    if (i >= GO_ID && v > lamax) { lamax = v; laidx = i; }
  }
  s_max[tid] = lmax; s_amax[tid] = lamax; s_aidx[tid] = laidx;
  __syncthreads();
  for (int s = 128; s > 0; s >>= 1) {
    if (tid < s) {
      s_max[tid] = fmaxf(s_max[tid], s_max[tid + s]);
      float ov = s_amax[tid + s]; int oi = s_aidx[tid + s];
      if (ov > s_amax[tid] || (ov == s_amax[tid] && oi < s_aidx[tid])) {
        s_amax[tid] = ov; s_aidx[tid] = oi;
      }
    }
    __syncthreads();
  }
  const float gmax = s_max[0];
  const int   tok  = s_aidx[0];
  __syncthreads();

  // Pass 2: sum of exp.
  float lsum = 0.0f;
  for (int i = tid; i < VOCAB; i += 256) lsum += expf(row[i] - gmax);
  s_max[tid] = lsum;
  __syncthreads();
  for (int s = 128; s > 0; s >>= 1) {
    if (tid < s) s_max[tid] += s_max[tid + s];
    __syncthreads();
  }
  const float inv = 1.0f / s_max[0];

  // Pass 3: write probabilities (never masked in reference).
  for (int i = tid; i < VOCAB; i += 256)
    w_pro_out[(size_t)b * VOCAB + i] = expf(row[i] - gmax) * inv;

  // Token, flag (alive uses flag BEFORE update, per reference).
  __shared__ float s_alive;
  if (tid == 0) {
    int f = flags[b];
    float alive = f ? 0.0f : 1.0f;
    s_alive = alive;
    w_o_out[b]   = alive * (float)tok;
    alive_out[b] = f ? 0 : 1;
    flags[b] = f | (tok == EOS_ID);
  }
  __syncthreads();
  const float alive = s_alive;

  // Embedding gather: next_emb unmasked (carry), emb_seq masked (output).
  if (tid < EMBED) {
    float e = emb_table[(size_t)tok * EMBED + tid];
    next_emb[b * EMBED + tid]    = e;
    emb_seq_out[b * EMBED + tid] = alive * e;
  }
}

// length[b] = sum_t alive[t][b]
__global__ void k_length(const int* __restrict__ alive, float* __restrict__ len_out) {
  int b = threadIdx.x;
  if (b < BATCH) {
    int s = 0;
    for (int t = 0; t < MAX_LEN; ++t) s += alive[t * BATCH + b];
    len_out[b] = (float)s;
  }
}

// ---------------------------------------------------------------------------
// Host side
// ---------------------------------------------------------------------------
extern "C" void kernel_launch(void* const* d_in, const int* in_sizes, int n_in,
                              void* d_out, int out_size, void* d_ws, size_t ws_size,
                              hipStream_t stream) {
  (void)in_sizes; (void)n_in; (void)out_size; (void)ws_size;

  const float* emb_table = (const float*)d_in[0];  // [32000,256]
  const float* W_ih      = (const float*)d_in[1];  // [1536,256]
  const float* b_ih      = (const float*)d_in[2];  // [1536]
  const float* W_hh      = (const float*)d_in[3];  // [1536,512]
  const float* b_hh      = (const float*)d_in[4];  // [1536]
  const float* W_out     = (const float*)d_in[5];  // [32000,512]
  const float* b_out     = (const float*)d_in[6];  // [32000]
  const float* h0        = (const float*)d_in[7];  // [32,512]

  float* out = (float*)d_out;
  char*  ws  = (char*)d_ws;

  // Workspace layout (256B aligned)
  const size_t o_Wih16  = 0;                                   // 1536*256*2
  const size_t o_Whh16  = o_Wih16  + (size_t)1536 * 256 * 2;   // 1536*512*2
  const size_t o_Wout16 = o_Whh16  + (size_t)1536 * 512 * 2;   // 32000*512*2
  const size_t o_h0     = o_Wout16 + (size_t)VOCAB * HIDDEN * 2;
  const size_t o_h1     = o_h0 + (size_t)BATCH * HIDDEN * 4;
  const size_t o_e0     = o_h1 + (size_t)BATCH * HIDDEN * 4;
  const size_t o_e1     = o_e0 + (size_t)BATCH * EMBED * 4;
  const size_t o_logits = o_e1 + (size_t)BATCH * EMBED * 4;
  const size_t o_flags  = o_logits + (size_t)BATCH * VOCAB * 4;
  const size_t o_alive  = o_flags + 256;

  _Float16* Wih16  = (_Float16*)(ws + o_Wih16);
  _Float16* Whh16  = (_Float16*)(ws + o_Whh16);
  _Float16* Wout16 = (_Float16*)(ws + o_Wout16);
  float* hbuf[2]   = {(float*)(ws + o_h0), (float*)(ws + o_h1)};
  float* ebuf[2]   = {(float*)(ws + o_e0), (float*)(ws + o_e1)};
  float* logits    = (float*)(ws + o_logits);
  int*   flags     = (int*)(ws + o_flags);
  int*   alive     = (int*)(ws + o_alive);

  // d_out layout: w_pro [64,32,32000] | w_o [64,32] | emb_seq [64,32,256] | length [32]
  float* out_wpro = out;
  float* out_wo   = out + (size_t)MAX_LEN * BATCH * VOCAB;
  float* out_emb  = out_wo + (size_t)MAX_LEN * BATCH;
  float* out_len  = out_emb + (size_t)MAX_LEN * BATCH * EMBED;

  // ---- one-time prep: f16 weights + initial state ----
  k_f32_to_f16<<<1024, 256, 0, stream>>>(W_ih,  Wih16,  1536 * 256);
  k_f32_to_f16<<<1024, 256, 0, stream>>>(W_hh,  Whh16,  1536 * 512);
  k_f32_to_f16<<<4096, 256, 0, stream>>>(W_out, Wout16, VOCAB * HIDDEN);
  k_init_state<<<(BATCH * HIDDEN + 255) / 256, 256, 0, stream>>>(
      h0, emb_table, hbuf[0], ebuf[0], flags);

  // ---- 64 sequential decode steps ----
  for (int t = 0; t < MAX_LEN; ++t) {
    const int in  = t & 1;
    const int oux = (t + 1) & 1;

    k_gru_cell<<<HIDDEN / 16, 64, 0, stream>>>(
        ebuf[in], hbuf[in], Wih16, Whh16, b_ih, b_hh, hbuf[oux]);

    k_logits<<<VOCAB / 64, 64, 0, stream>>>(
        hbuf[oux], Wout16, b_out, logits);

    k_softmax_argmax<<<BATCH, 256, 0, stream>>>(
        logits, emb_table,
        out_wpro + (size_t)t * BATCH * VOCAB,
        out_wo   + (size_t)t * BATCH,
        out_emb  + (size_t)t * BATCH * EMBED,
        ebuf[oux], flags, alive + t * BATCH);
  }

  k_length<<<1, 32, 0, stream>>>(alive, out_len);
}
